// RGLRU_16904991277207
// MI455X (gfx1250) — compile-verified, matrix operand order checked by reading, not run on hardware
//
#include <hip/hip_runtime.h>
#include <hip/hip_bf16.h>
#include <math.h>

typedef __bf16 bf16_t;
typedef __attribute__((ext_vector_type(16))) __bf16 v16bf;
typedef __attribute__((ext_vector_type(8)))  __bf16 v8bf;
typedef __attribute__((ext_vector_type(8)))  float  v8f;

#define RG_B   4
#define RG_L   4096
#define RG_D   2048
#define RG_H   8
#define RG_BW  256                 // block width D/H
#define RG_LC  128                 // chunk length (time)
#define RG_NC  (RG_L / RG_LC)      // 32 chunks
#define RG_MS  32                  // timestep sub-tile (2 WMMA M-tiles)
#define RG_NSUB (RG_LC / RG_MS)    // 4 sub-tiles per chunk
#define RG_KT  (RG_BW / 32)        // 8 K-steps of 32 (bf16 WMMA K)

// ---------------------------------------------------------------------------
// Prep: transpose weights to [gate][h][n][k] bf16 (B-matrix friendly, K
// contiguous per output row) and precompute c8[d] = 8*softplus(a_param[d]).
// ---------------------------------------------------------------------------
__global__ void rglru_prep(const float* __restrict__ ig_w,
                           const float* __restrict__ ag_w,
                           const float* __restrict__ a_param,
                           bf16_t* __restrict__ wt,
                           float* __restrict__ c8)
{
  const int idx = blockIdx.x * blockDim.x + threadIdx.x;
  const int per_gate = RG_H * RG_BW * RG_BW;
  if (idx < 2 * per_gate) {
    const int g = idx / per_gate;
    const int r = idx - g * per_gate;
    const int h = r / (RG_BW * RG_BW);
    const int e = r - h * (RG_BW * RG_BW);
    const int n = e / RG_BW;            // output channel j
    const int k = e - n * RG_BW;        // input  channel i
    const float* src = g ? ag_w : ig_w; // w[h][i][j]
    wt[idx] = (bf16_t)src[(h * RG_BW + k) * RG_BW + n];
  }
  if (idx < RG_D) {
    const float a = a_param[idx];
    const float sp = (a > 20.0f) ? a : log1pf(__expf(a));
    c8[idx] = 8.0f * sp;
  }
}

// ---------------------------------------------------------------------------
// Fused kernel: per (chunk c, head h, batch b) block.
//  - WMMA bf16 GEMM for both gates on 32-timestep sub-tiles
//  - elementwise gate math + local scan (h starts at 0 for every chunk)
//  - emits y_local, a (bf16), per-chunk decay product P, local end state Hloc
// 112 KB LDS, VGPRs bounded for 2 blocks/WGP so stage/GEMM/scan phases of
// different blocks overlap on the VMEM, WMMA and VALU pipes.
// ---------------------------------------------------------------------------
__launch_bounds__(256, 2)
__global__ void rglru_gemm_scan(const float* __restrict__ x,
                                const bf16_t* __restrict__ wt,
                                const float* __restrict__ ig_b,
                                const float* __restrict__ ag_b,
                                const float* __restrict__ c8,
                                float* __restrict__ y,
                                bf16_t* __restrict__ a_out,
                                float* __restrict__ P,
                                float* __restrict__ Hloc)
{
  __shared__ __align__(16) float  sXf[RG_MS][RG_BW];    // 32 KB
  __shared__ __align__(16) bf16_t sXh[RG_MS][RG_BW];    // 16 KB
  __shared__ __align__(16) float  sG[2][RG_MS][RG_BW];  // 64 KB

  const int c   = blockIdx.x;
  const int h   = blockIdx.y;
  const int b   = blockIdx.z;
  const int tid = threadIdx.x;
  const int wave = tid >> 5;
  const int lane = tid & 31;
  const int lh   = lane & 15;   // M / N row within tile
  const int g16  = lane >> 4;   // lane-half selects K sub-range / M upper half

  const int d    = tid;              // channel within head (0..255)
  const int gch  = h * RG_BW + d;    // global channel
  const float bias_i = ig_b[gch];
  const float bias_a = ag_b[gch];
  const float c8v    = c8[gch];

  float hreg = 0.0f;   // local scan state (chunk starts at 0)
  float cum  = 1.0f;   // running product of a over the chunk

  const float* xbase = x + ((size_t)b * RG_L + (size_t)c * RG_LC) * RG_D + h * RG_BW;

  for (int sub = 0; sub < RG_NSUB; ++sub) {
    const int t0 = c * RG_LC + sub * RG_MS;
    const float* xsrc = xbase + (size_t)sub * RG_MS * RG_D;

    // Launder the weight base so B-fragment loads cannot be hoisted out of
    // the sub loop (keeps VGPRs < 256 -> 2 blocks/WGP; weights stay in L2).
    unsigned woff = 0;
    asm volatile("" : "+v"(woff));
    const bf16_t* wts = wt + woff;

    // ---- stage X sub-tile into LDS (f32 for elementwise, bf16 for WMMA A)
    #pragma unroll
    for (int i = tid; i < RG_MS * RG_BW; i += 256) {
      const int r  = i >> 8;
      const int cc = i & 255;
      const float v = xsrc[(size_t)r * RG_D + cc];
      sXf[r][cc] = v;
      sXh[r][cc] = (bf16_t)v;
    }
    // prefetch next sub-tile (one 128B line per segment of each row)
    if (sub + 1 < RG_NSUB) {
      const int pr = tid >> 3, ps = tid & 7;
      __builtin_prefetch(xsrc + (size_t)(RG_MS + pr) * RG_D + ps * 32, 0, 0);
    }
    __syncthreads();

    // ---- GEMM: 2 gates x 16 n-tiles x 2 m-tiles = 64 tiles, 8 per wave
    v8f acc[4][2] = {};
    #pragma unroll
    for (int kk = 0; kk < RG_KT; ++kk) {
      const int kbase = kk * 32 + g16 * 8;
      // A fragments (two M-tiles): lane lh row M=lh; elem j -> K=(j/8)*16+g16*8+j%8
      v16bf afrag[2];
      #pragma unroll
      for (int mt = 0; mt < 2; ++mt) {
        const v8bf alo = *(const v8bf*)&sXh[mt * 16 + lh][kbase];
        const v8bf ahi = *(const v8bf*)&sXh[mt * 16 + lh][kbase + 16];
        afrag[mt] = __builtin_shufflevector(alo, ahi,
            0,1,2,3,4,5,6,7,8,9,10,11,12,13,14,15);
      }
      #pragma unroll
      for (int q = 0; q < 4; ++q) {
        const int tt   = wave * 4 + q;
        const int gate = tt >> 4;
        const int nt   = tt & 15;
        const bf16_t* brow = wts
            + ((size_t)(gate * RG_H + h) * RG_BW + nt * 16 + lh) * RG_BW + kbase;
        const v8bf blo = *(const v8bf*)brow;
        const v8bf bhi = *(const v8bf*)(brow + 16);
        const v16bf bfrag = __builtin_shufflevector(blo, bhi,
            0,1,2,3,4,5,6,7,8,9,10,11,12,13,14,15);
        acc[q][0] = __builtin_amdgcn_wmma_f32_16x16x32_bf16(
            false, afrag[0], false, bfrag, (short)0, acc[q][0], false, false);
        acc[q][1] = __builtin_amdgcn_wmma_f32_16x16x32_bf16(
            false, afrag[1], false, bfrag, (short)0, acc[q][1], false, false);
      }
    }
    // D layout: lane lh col N=lh; VGPR v holds M = v + 8*g16
    #pragma unroll
    for (int q = 0; q < 4; ++q) {
      const int tt   = wave * 4 + q;
      const int gate = tt >> 4;
      const int nt   = tt & 15;
      #pragma unroll
      for (int mt = 0; mt < 2; ++mt)
        #pragma unroll
        for (int v = 0; v < 8; ++v)
          sG[gate][mt * 16 + v + 8 * g16][nt * 16 + lh] = acc[q][mt][v];
    }
    __syncthreads();

    // ---- elementwise + local recurrence, one thread per channel
    const size_t orow = ((size_t)b * RG_L + t0) * RG_D + gch;
    #pragma unroll 8
    for (int t = 0; t < RG_MS; ++t) {
      const float xv = sXf[t][d];
      const float gx = sG[0][t][d] + bias_i;
      const float ga = sG[1][t][d] + bias_a;
      const float sgx = 1.0f / (1.0f + __expf(-gx));
      const float sga = 1.0f / (1.0f + __expf(-ga));
      const float log_a = -sga * c8v;
      const float av = __expf(log_a);
      const float mult = sqrtf(fmaxf(1.0f - av * av, 0.0f)); // sqrt(1-exp(2 log_a))
      const float nx = xv * sgx * mult;
      hreg = av * hreg + nx;
      cum *= av;
      y[orow + (size_t)t * RG_D]     = hreg;
      a_out[orow + (size_t)t * RG_D] = (bf16_t)av;
    }
    __syncthreads();
  }

  const size_t pidx = ((size_t)b * RG_NC + c) * RG_D + gch;
  P[pidx]    = cum;
  Hloc[pidx] = hreg;
}

// ---------------------------------------------------------------------------
// Serial chunk-boundary combine: per channel, walk the 32 chunks.
// delta[c] = true incoming h for chunk c (local scans used 0).
// ---------------------------------------------------------------------------
__global__ void rglru_combine(const float* __restrict__ hidden,
                              const float* __restrict__ P,
                              const float* __restrict__ Hloc,
                              float* __restrict__ delta,
                              float* __restrict__ hfinal)
{
  const int idx = blockIdx.x * blockDim.x + threadIdx.x;
  if (idx >= RG_B * RG_D) return;
  const int b  = idx / RG_D;
  const int ch = idx - b * RG_D;
  float v = hidden[idx];
  for (int c = 0; c < RG_NC; ++c) {
    const size_t k = ((size_t)b * RG_NC + c) * RG_D + ch;
    delta[k] = v;
    v = Hloc[k] + P[k] * v;
  }
  hfinal[idx] = v;
}

// ---------------------------------------------------------------------------
// Correction: y[t] += cumprod(a[0..t]) * delta  within each chunk.
// Skips entirely when delta == 0 (e.g. chunk 0 with zero hidden_state).
// ---------------------------------------------------------------------------
__launch_bounds__(256)
__global__ void rglru_fixup(const float* __restrict__ delta,
                            const bf16_t* __restrict__ a_s,
                            float* __restrict__ y)
{
  const int c = blockIdx.x, h = blockIdx.y, b = blockIdx.z;
  const int gch = h * RG_BW + threadIdx.x;
  const float dlt = delta[((size_t)b * RG_NC + c) * RG_D + gch];
  if (dlt == 0.0f) return;
  size_t base = ((size_t)b * RG_L + (size_t)c * RG_LC) * RG_D + gch;
  float cumv = dlt;
  #pragma unroll 4
  for (int t = 0; t < RG_LC; ++t) {
    const float av = (float)a_s[base];
    cumv *= av;
    y[base] += cumv;
    base += RG_D;
  }
}

// ---------------------------------------------------------------------------
extern "C" void kernel_launch(void* const* d_in, const int* in_sizes, int n_in,
                              void* d_out, int out_size, void* d_ws, size_t ws_size,
                              hipStream_t stream)
{
  (void)in_sizes; (void)n_in; (void)out_size; (void)ws_size;
  const float* x       = (const float*)d_in[0];
  const float* hidden  = (const float*)d_in[1];
  const float* a_param = (const float*)d_in[2];
  const float* ig_w    = (const float*)d_in[3];
  const float* ig_b    = (const float*)d_in[4];
  const float* ag_w    = (const float*)d_in[5];
  const float* ag_b    = (const float*)d_in[6];

  float* y      = (float*)d_out;
  float* hfinal = y + (size_t)RG_B * RG_L * RG_D;   // tuple: (y, h_final)

  // workspace carve-out (all regions fully rewritten each call)
  char* ws = (char*)d_ws;
  auto al = [](size_t v) { return (v + 255) & ~(size_t)255; };
  const size_t szWt = (size_t)2 * RG_H * RG_BW * RG_BW * sizeof(bf16_t); // 2 MB
  const size_t szC8 = (size_t)RG_D * sizeof(float);
  const size_t szPC = (size_t)RG_B * RG_NC * RG_D * sizeof(float);      // 1 MB
  const size_t szA  = (size_t)RG_B * RG_L * RG_D * sizeof(bf16_t);      // 64 MB
  size_t off = 0;
  bf16_t* wt   = (bf16_t*)(ws + off); off += al(szWt);
  float*  c8   = (float*) (ws + off); off += al(szC8);
  float*  Pp   = (float*) (ws + off); off += al(szPC);
  float*  Hl   = (float*) (ws + off); off += al(szPC);
  float*  dlt  = (float*) (ws + off); off += al(szPC);
  bf16_t* a_s  = (bf16_t*)(ws + off); off += al(szA);

  const int prepN = 2 * RG_H * RG_BW * RG_BW;
  rglru_prep<<<dim3((prepN + 255) / 256), dim3(256), 0, stream>>>(
      ig_w, ag_w, a_param, wt, c8);

  rglru_gemm_scan<<<dim3(RG_NC, RG_H, RG_B), dim3(256), 0, stream>>>(
      x, wt, ig_b, ag_b, c8, y, a_s, Pp, Hl);

  rglru_combine<<<dim3((RG_B * RG_D + 255) / 256), dim3(256), 0, stream>>>(
      hidden, Pp, Hl, dlt, hfinal);

  rglru_fixup<<<dim3(RG_NC, RG_H, RG_B), dim3(256), 0, stream>>>(
      dlt, a_s, y);
}
